// VectorQuantize_78391743087177
// MI455X (gfx1250) — matrix operand order repaired and checked by reading, not compiled, not run on hardware
//
#include <hip/hip_runtime.h>
#include <hip/hip_bf16.h>
#include <float.h>

typedef __attribute__((ext_vector_type(16))) __bf16 v16bf;
typedef __attribute__((ext_vector_type(8)))  float  v8f;

#define DECAYF 0.99f
#define EPSF   1e-5f

static constexpr int Bb = 8, Tt = 2048, Dd = 256, Vv = 8192;
static constexpr int NT = Bb * Tt;          // 16384 tokens
static constexpr int TOK_PER_BLOCK = 128;   // 4 waves * 32 tokens
static constexpr int STAGE = 64;            // codes per LDS stage (32KB bf16)

union Frag {
    v16bf v;
    float4 f4[2];
};

// ---------------- prep kernels ----------------

__global__ void vq_zero_kernel(float* __restrict__ counts, float* __restrict__ esum) {
    int i = blockIdx.x * blockDim.x + threadIdx.x;
    if (i < Vv) counts[i] = 0.0f;
    if (i < Vv * Dd) esum[i] = 0.0f;
}

__global__ void vq_cvt_input_kernel(const float* __restrict__ x, __bf16* __restrict__ xb) {
    int i = blockIdx.x * blockDim.x + threadIdx.x;   // NT*Dd total, exact multiple
    xb[i] = (__bf16)x[i];
}

__global__ __launch_bounds__(256)
void vq_cvt_embed_kernel(const float* __restrict__ e, __bf16* __restrict__ eb,
                         float* __restrict__ enorm) {
    __shared__ float red[256];
    int v = blockIdx.x;
    int d = threadIdx.x;
    size_t idx = (size_t)v * Dd + d;
    float val = e[idx];
    eb[idx] = (__bf16)val;
    red[d] = val * val;
    __syncthreads();
    #pragma unroll
    for (int s = 128; s > 0; s >>= 1) {
        if (d < s) red[d] += red[d + s];
        __syncthreads();
    }
    if (d == 0) enorm[v] = red[0];
}

// ---------------- async global -> LDS staging (CDNA5 path) ----------------
// GVS mode: addr = SGPR64 + VGPR32 + IOFFSET; IOFFSET is added to BOTH the
// global and the LDS address, so one (lds_addr, voffset) pair covers all 16
// b128 transfers of a thread's 256-byte slice. Tracked by ASYNCcnt.

#define ASYNC_CP_B128(ldsv, voffv, sbase, OFF)                                 \
    asm volatile("global_load_async_to_lds_b128 %0, %1, %2 offset:" #OFF       \
                 :: "v"(ldsv), "v"(voffv), "s"(sbase) : "memory")

__device__ __forceinline__ void async_stage_32k(unsigned lds, unsigned voff,
                                                const __bf16* sbase) {
    // 128 threads * 16B, stride 2048B, 16 steps = 32KB
    ASYNC_CP_B128(lds, voff, sbase, 0);
    ASYNC_CP_B128(lds, voff, sbase, 2048);
    ASYNC_CP_B128(lds, voff, sbase, 4096);
    ASYNC_CP_B128(lds, voff, sbase, 6144);
    ASYNC_CP_B128(lds, voff, sbase, 8192);
    ASYNC_CP_B128(lds, voff, sbase, 10240);
    ASYNC_CP_B128(lds, voff, sbase, 12288);
    ASYNC_CP_B128(lds, voff, sbase, 14336);
    ASYNC_CP_B128(lds, voff, sbase, 16384);
    ASYNC_CP_B128(lds, voff, sbase, 18432);
    ASYNC_CP_B128(lds, voff, sbase, 20480);
    ASYNC_CP_B128(lds, voff, sbase, 22528);
    ASYNC_CP_B128(lds, voff, sbase, 24576);
    ASYNC_CP_B128(lds, voff, sbase, 26624);
    ASYNC_CP_B128(lds, voff, sbase, 28672);
    ASYNC_CP_B128(lds, voff, sbase, 30720);
}

__device__ __forceinline__ void wait_async0() {
    asm volatile("s_wait_asynccnt 0x0" ::: "memory");
}

// ---------------- fused WMMA distance + argmin ----------------
// grid: NT/128 blocks, 128 threads (4 waves). Wave w owns 32 tokens (two
// 16-row M-tiles) so every B fragment feeds 2 WMMAs -> half the LDS traffic
// per matrix op. Scans all V codes; LDS double-buffered 64-code stages are
// filled by the async engine while WMMAs run on the other buffer.

__global__ __launch_bounds__(128)
void vq_argmin_kernel(const __bf16* __restrict__ Xb, const __bf16* __restrict__ Eb,
                      const float* __restrict__ enorm, int* __restrict__ codes) {
    __shared__ __bf16 sB[2 * STAGE * Dd];   // 2 x 32 KB double buffer

    const int tid  = threadIdx.x;
    const int wave = tid >> 5;
    const int lane = tid & 31;
    const int half = lane >> 4;   // 0: lanes 0-15, 1: lanes 16-31
    const int l16  = lane & 15;

    const int tok_base = blockIdx.x * TOK_PER_BLOCK + wave * 32;

    // A fragments: 2 M-tiles x 16 tokens x K=256 bf16, resident in VGPRs.
    // ISA 16-bit A 16x32 layout: lane half 0 -> K {k0+0..7, k0+16..23},
    // lane half 1 -> K {k0+8..15, k0+24..31}; M = l16.
    Frag a0[8], a1[8];
    {
        const __bf16* arow0 = Xb + (size_t)(tok_base + l16) * Dd;
        const __bf16* arow1 = arow0 + (size_t)16 * Dd;
        #pragma unroll
        for (int kf = 0; kf < 8; ++kf) {
            int k = kf * 32 + half * 8;
            a0[kf].f4[0] = *(const float4*)(arow0 + k);
            a0[kf].f4[1] = *(const float4*)(arow0 + k + 16);
            a1[kf].f4[0] = *(const float4*)(arow1 + k);
            a1[kf].f4[1] = *(const float4*)(arow1 + k + 16);
        }
    }

    float bestV[2][8];
    int   bestI[2][8];
    #pragma unroll
    for (int m = 0; m < 2; ++m)
        #pragma unroll
        for (int r = 0; r < 8; ++r) { bestV[m][r] = FLT_MAX; bestI[m][r] = 0; }

    // LDS byte addresses of the two buffers for this thread's slice
    const unsigned byte_off = (unsigned)tid * 16u;
    const unsigned lds_base = (unsigned)(size_t)(&sB[0]);
    const unsigned lds0 = lds_base + byte_off;
    const unsigned lds1 = lds_base + (unsigned)(STAGE * Dd * 2) + byte_off;

    // prologue: stage 0 -> buffer 0
    async_stage_32k(lds0, byte_off, Eb);
    wait_async0();
    __syncthreads();

    const int NSTAGES = Vv / STAGE;   // 128
    for (int i = 0; i < NSTAGES; ++i) {
        const int s = i * STAGE;
        // kick off next stage into the other buffer (overlaps the WMMAs)
        if (i + 1 < NSTAGES) {
            const __bf16* gsrc = Eb + (size_t)(s + STAGE) * Dd;
            async_stage_32k(((i + 1) & 1) ? lds1 : lds0, byte_off, gsrc);
        }

        const __bf16* sbuf = sB + (i & 1) * (STAGE * Dd);

        #pragma unroll
        for (int nt = 0; nt < STAGE / 16; ++nt) {
            // B (K x N) bf16 32x16 layout: lane half h holds
            // K {k0+h*8..+7, k0+16+h*8..+7} for column N = l16.
            const __bf16* brow = sbuf + (size_t)(nt * 16 + l16) * Dd;
            v8f c0 = {};
            v8f c1 = {};
            #pragma unroll
            for (int kf = 0; kf < 8; ++kf) {
                Frag b;
                int k = kf * 32 + half * 8;
                b.f4[0] = *(const float4*)(brow + k);
                b.f4[1] = *(const float4*)(brow + k + 16);
                c0 = __builtin_amdgcn_wmma_f32_16x16x32_bf16(
                        false, a0[kf].v, false, b.v, (short)0, c0, false, false);
                c1 = __builtin_amdgcn_wmma_f32_16x16x32_bf16(
                        false, a1[kf].v, false, b.v, (short)0, c1, false, false);
            }
            // score = ||e||^2 - 2 x.e  (||x||^2 constant per token, dropped)
            float en  = enorm[s + nt * 16 + l16];
            int   idx = s + nt * 16 + l16;
            #pragma unroll
            for (int r = 0; r < 8; ++r) {
                float s0 = __builtin_fmaf(-2.0f, c0[r], en);
                float s1 = __builtin_fmaf(-2.0f, c1[r], en);
                if (s0 < bestV[0][r]) { bestV[0][r] = s0; bestI[0][r] = idx; }
                if (s1 < bestV[1][r]) { bestV[1][r] = s1; bestI[1][r] = idx; }
            }
        }

        wait_async0();      // this wave's copy of stage i+1 is done
        __syncthreads();    // everyone done computing buf[i&1] + copies visible
    }

    // C layout: VGPR r, lanes 0-15 -> M=r ; lanes 16-31 -> M=8+r.
    // Reduce argmin across the 16 lanes of each half (first-index tie-break).
    #pragma unroll
    for (int m = 0; m < 2; ++m) {
        #pragma unroll
        for (int r = 0; r < 8; ++r) {
            float v  = bestV[m][r];
            int   bi = bestI[m][r];
            #pragma unroll
            for (int off = 8; off > 0; off >>= 1) {
                float ov = __shfl_xor(v, off, 16);
                int   oi = __shfl_xor(bi, off, 16);
                if (ov < v || (ov == v && oi < bi)) { v = ov; bi = oi; }
            }
            if (l16 == 0)
                codes[tok_base + m * 16 + half * 8 + r] = bi;
        }
    }
}

// ---------------- gather + EMA accumulation ----------------

__global__ __launch_bounds__(256)
void vq_gather_ema_kernel(const float* __restrict__ x, const float* __restrict__ embed,
                          const int* __restrict__ codes, float* __restrict__ quant_out,
                          float* __restrict__ codes_out, float* __restrict__ counts,
                          float* __restrict__ esum) {
    int t = blockIdx.x;
    int d = threadIdx.x;
    int code = codes[t];
    size_t ti = (size_t)t * Dd + d;
    size_t ci = (size_t)code * Dd + d;
    // straight-through: quantize value == gathered embed row (exact f32)
    quant_out[ti] = embed[ci];
    atomicAdd(&esum[ci], x[ti]);
    if (d == 0) {
        atomicAdd(&counts[code], 1.0f);
        codes_out[t] = (float)code;
    }
}

__global__ void vq_finalize_kernel(const float* __restrict__ cs, const float* __restrict__ eavg,
                                   const float* __restrict__ counts, const float* __restrict__ esum,
                                   float* __restrict__ new_embed, float* __restrict__ new_cs,
                                   float* __restrict__ new_avg) {
    int i = blockIdx.x * blockDim.x + threadIdx.x;
    if (i >= Vv * Dd) return;
    int v = i >> 8;  // / Dd
    float ncs = cs[v] * DECAYF + counts[v] * (1.0f - DECAYF);
    float nea = eavg[i] * DECAYF + (esum[i] * (1.0f / Tt)) * (1.0f - DECAYF);
    new_avg[i]   = nea;
    new_embed[i] = nea / (ncs + EPSF);
    if ((i & (Dd - 1)) == 0) new_cs[v] = ncs;
}

// ---------------- launch ----------------

extern "C" void kernel_launch(void* const* d_in, const int* in_sizes, int n_in,
                              void* d_out, int out_size, void* d_ws, size_t ws_size,
                              hipStream_t stream) {
    const float* x     = (const float*)d_in[0];   // [NT, D]
    const float* embed = (const float*)d_in[1];   // [V, D]
    const float* cs    = (const float*)d_in[2];   // [V]
    const float* eavg  = (const float*)d_in[3];   // [V, D]

    // workspace layout
    char* ws = (char*)d_ws;
    size_t off = 0;
    __bf16* Xb    = (__bf16*)(ws + off); off += (size_t)NT * Dd * 2;   // 8 MB
    __bf16* Eb    = (__bf16*)(ws + off); off += (size_t)Vv * Dd * 2;   // 4 MB
    float*  enorm = (float*)(ws + off);  off += (size_t)Vv * 4;
    float*  counts= (float*)(ws + off);  off += (size_t)Vv * 4;
    float*  esum  = (float*)(ws + off);  off += (size_t)Vv * Dd * 4;   // 8 MB
    int*    codes = (int*)(ws + off);    off += (size_t)NT * 4;

    // outputs, flat in return order
    float* out       = (float*)d_out;
    float* q_out     = out;                               // NT*D
    float* codes_out = q_out + (size_t)NT * Dd;           // NT
    float* embed_out = codes_out + NT;                    // V*D
    float* cs_out    = embed_out + (size_t)Vv * Dd;       // V
    float* avg_out   = cs_out + Vv;                       // V*D

    vq_zero_kernel<<<(Vv * Dd + 255) / 256, 256, 0, stream>>>(counts, esum);
    vq_cvt_input_kernel<<<(NT * Dd) / 256, 256, 0, stream>>>(x, Xb);
    vq_cvt_embed_kernel<<<Vv, 256, 0, stream>>>(embed, Eb, enorm);

    vq_argmin_kernel<<<NT / TOK_PER_BLOCK, 128, 0, stream>>>(Xb, Eb, enorm, codes);

    vq_gather_ema_kernel<<<NT, Dd, 0, stream>>>(x, embed, codes, q_out, codes_out,
                                                counts, esum);
    vq_finalize_kernel<<<(Vv * Dd + 255) / 256, 256, 0, stream>>>(
        cs, eavg, counts, esum, embed_out, cs_out, avg_out);
}
